// GNNEncoder_66408784331090
// MI455X (gfx1250) — matrix operand order repaired and checked by reading, not compiled, not run on hardware
//
#include <hip/hip_runtime.h>

typedef __bf16 bf16_t;
typedef __attribute__((ext_vector_type(16))) __bf16 v16bf;
typedef __attribute__((ext_vector_type(8)))  __bf16 v8bf;
typedef __attribute__((ext_vector_type(8)))  float  v8f;

#define D 128

// ---------------------------------------------------------------- helpers

// Branchless erf (Abramowitz-Stegun 7.1.26, |err| < 1.5e-7): one v_exp, one
// v_rcp, 6 FMAs, copysign -- no EXEC divergence, unlike libm erff.
static __device__ __forceinline__ float erf_fast(float x) {
  float z = __builtin_fabsf(x);
  float t = __builtin_amdgcn_rcpf(__builtin_fmaf(0.3275911f, z, 1.0f));
  float p = __builtin_fmaf(t, 1.061405429f, -1.453152027f);
  p = __builtin_fmaf(t, p, 1.421413741f);
  p = __builtin_fmaf(t, p, -0.284496736f);
  p = __builtin_fmaf(t, p, 0.254829592f);
  float e = __expf(-z * z);
  float r = __builtin_fmaf(-p * t, e, 1.0f);
  return __builtin_copysignf(r, x);
}

static __device__ __forceinline__ float gelu_f(float x) {
  return 0.5f * x * (1.0f + erf_fast(x * 0.70710678118654752440f));
}

// float atomic max via monotone integer trick (correct for all finite floats)
static __device__ __forceinline__ void atomicMaxF(float* addr, float v) {
  if (v >= 0.0f) atomicMax((int*)addr, __float_as_int(v));
  else           atomicMin((unsigned int*)addr, __float_as_uint(v));
}

// B fragment for v_wmma_f32_16x16x32_bf16:  B[k,n] = W[n,k]  (out = A @ W^T)
// lane: n = ntile*16 + (lane&15); j=0..7 -> k = kb + j ; j=8..15 -> k = kb + 16 + (j-8), kb = kstep*32 + ((lane>>4)&1)*8
static __device__ __forceinline__ v16bf wmma_b_frag(const bf16_t* sW, int lane, int nt, int ks) {
  int n  = nt * 16 + (lane & 15);
  int kb = ks * 32 + ((lane >> 4) & 1) * 8;
  v8bf b0 = *(const v8bf*)(sW + n * D + kb);
  v8bf b1 = *(const v8bf*)(sW + n * D + kb + 16);
  return __builtin_shufflevector(b0, b1, 0, 1, 2, 3, 4, 5, 6, 7, 8, 9, 10, 11, 12, 13, 14, 15);
}

// A fragment from an fp32 row (lane's row = tile row (lane&15)), with optional
// gelu and scale applied before bf16 conversion.
// ISA 16-bit A layout: j=0..7 -> k = kbase + j ; j=8..15 -> k = kbase + 16 + (j-8)
// where kbase = kstep*32 + h*8, h = lane>>4.
static __device__ __forceinline__ v16bf a_frag_f32(const float* row, int kbase, float scale, int dg) {
  v8f x0 = *(const v8f*)(row + kbase);
  v8f x1 = *(const v8f*)(row + kbase + 16);
  v16bf a;
#pragma unroll
  for (int j = 0; j < 8; ++j) {
    float u = x0[j], v = x1[j];
    if (dg) { u = gelu_f(u); v = gelu_f(v); }
    a[j]     = (bf16_t)(u * scale);
    a[j + 8] = (bf16_t)(v * scale);
  }
  return a;
}

#define WMMA_BF16(A, B, C) \
  __builtin_amdgcn_wmma_f32_16x16x32_bf16(false, (A), false, (B), (short)0, (C), false, false)

// ---------------------------------------------------------------- small prep kernels

// Fold/convert weights to bf16 once; build combined bias for P.
__global__ void k_prep(const float* __restrict__ Wasrc, const float* __restrict__ Wadst,
                       const float* __restrict__ Wasub, const float* __restrict__ Wamul,
                       const float* __restrict__ Wpool, const float* __restrict__ Wpool2,
                       const float* __restrict__ Wself, const float* __restrict__ Wneigh,
                       const float* __restrict__ Wneigh2, const float* __restrict__ Wmlp,
                       const float* __restrict__ basrc, const float* __restrict__ badst,
                       const float* __restrict__ basub, const float* __restrict__ bamul,
                       bf16_t* __restrict__ wts, float* __restrict__ biasP) {
  int i = blockIdx.x * blockDim.x + threadIdx.x;
  if (i < D * D) {
    wts[0 * D * D + i] = (bf16_t)(Wasub[i] + Wasrc[i]);   // -> P
    wts[1 * D * D + i] = (bf16_t)(Wadst[i] - Wasub[i]);   // -> Q
    wts[2 * D * D + i] = (bf16_t)Wpool[i];
    wts[3 * D * D + i] = (bf16_t)Wpool2[i];
    wts[4 * D * D + i] = (bf16_t)Wamul[i];
    wts[5 * D * D + i] = (bf16_t)Wself[i];
    wts[6 * D * D + i] = (bf16_t)Wneigh[i];
    wts[7 * D * D + i] = (bf16_t)Wneigh2[i];
    wts[8 * D * D + i] = (bf16_t)Wmlp[i];
    wts[9 * D * D + i] = (bf16_t)Wmlp[D * D + i];
    if (i < D) biasP[i] = basub[i] + bamul[i] + basrc[i] + badst[i];
  }
}

__global__ void k_featb(const float* __restrict__ f, bf16_t* __restrict__ o, int n) {
  int i = blockIdx.x * blockDim.x + threadIdx.x;
  if (i < n) o[i] = (bf16_t)f[i];
}

__global__ void k_init(float* __restrict__ nmax, float* __restrict__ nsum,
                       float* __restrict__ deg, int N) {
  int i = blockIdx.x * blockDim.x + threadIdx.x;
  if (i < N * D) { nmax[i] = -3.402823466e38f; nsum[i] = 0.0f; }
  if (i < N) deg[i] = 0.0f;
}

// ---------------------------------------------------------------- node GEMM: out = act(A @ W^T + b)

__global__ __launch_bounds__(256) void k_gemm_node(
    const float* __restrict__ A, const bf16_t* __restrict__ W,
    const float* __restrict__ bias, int do_gelu, int out_bf16,
    float* __restrict__ outf, bf16_t* __restrict__ outh, int N) {
  __shared__ __align__(32) bf16_t sW[D * D];
  {
    const uint4* s = (const uint4*)W;
    uint4* d = (uint4*)sW;
    for (int i = threadIdx.x; i < (D * D * 2) / 16; i += blockDim.x) d[i] = s[i];
  }
  __syncthreads();
  int wave = threadIdx.x >> 5, lane = threadIdx.x & 31;
  int h = (lane >> 4) & 1, ln = lane & 15;
  int row0 = (blockIdx.x * 8 + wave) * 16;
  if (row0 >= N) return;
  int arow = row0 + ln; if (arow > N - 1) arow = N - 1;
  const float* rowp = A + (size_t)arow * D;
  v16bf a[4];
#pragma unroll
  for (int ks = 0; ks < 4; ++ks) a[ks] = a_frag_f32(rowp, ks * 32 + h * 8, 1.0f, 0);
#pragma unroll
  for (int nt = 0; nt < 8; ++nt) {
    v8f acc = {};
#pragma unroll
    for (int ks = 0; ks < 4; ++ks)
      acc = WMMA_BF16(a[ks], wmma_b_frag(sW, lane, nt, ks), acc);
    int n = nt * 16 + ln;
    float bv = bias ? bias[n] : 0.0f;
#pragma unroll
    for (int r = 0; r < 8; ++r) {
      int row = row0 + r + 8 * h;
      if (row < N) {
        float v = acc[r] + bv;
        if (do_gelu) v = gelu_f(v);
        size_t idx = (size_t)row * D + n;
        if (out_bf16) outh[idx] = (bf16_t)v; else outf[idx] = v;
      }
    }
  }
}

// ---------------------------------------------------------------- edge kernel

__global__ __launch_bounds__(256) void k_edge(
    const bf16_t* __restrict__ featb, const int* __restrict__ src, const int* __restrict__ dst,
    const float* __restrict__ P, const float* __restrict__ Q,
    const bf16_t* __restrict__ Hp, const bf16_t* __restrict__ Hp2,
    const float* __restrict__ w_aout, const float* __restrict__ b_aout,
    const bf16_t* __restrict__ Wm,
    float* __restrict__ nmax, float* __restrict__ nsum, float* __restrict__ deg, int E) {
  __shared__ __align__(32) bf16_t sW[D * D];
  __shared__ int sS[8][16];
  __shared__ int sDd[8][16];
  {
    const uint4* s = (const uint4*)Wm;
    uint4* d = (uint4*)sW;
    for (int i = threadIdx.x; i < (D * D * 2) / 16; i += blockDim.x) d[i] = s[i];
  }
  int wave = threadIdx.x >> 5, lane = threadIdx.x & 31;
  int h = (lane >> 4) & 1, ln = lane & 15;
  int ebase = (blockIdx.x * 8 + wave) * 16;
  if (ebase < E && lane < 16) {
    int e = ebase + lane; if (e > E - 1) e = E - 1;
    sS[wave][lane]  = src[e];
    sDd[wave][lane] = dst[e];
  }
  __syncthreads();
  if (ebase >= E) return;                       // wave-uniform exit

  if (lane < 16 && (ebase + lane) < E) atomicAdd(&deg[sDd[wave][lane]], 1.0f);

  // ---- A = (fs * fd) in bf16, gathered via precomputed bf16 features
  int sm = sS[wave][ln], dm = sDd[wave][ln];
  const bf16_t* rs = featb + (size_t)sm * D;
  const bf16_t* rd = featb + (size_t)dm * D;
  v16bf a[4];
#pragma unroll
  for (int ks = 0; ks < 4; ++ks) {
    int kb = ks * 32 + h * 8;
    v8bf s0 = *(const v8bf*)(rs + kb);
    v8bf s1 = *(const v8bf*)(rs + kb + 16);
    v8bf d0 = *(const v8bf*)(rd + kb);
    v8bf d1 = *(const v8bf*)(rd + kb + 16);
    v16bf t;
#pragma unroll
    for (int j = 0; j < 8; ++j) {
      t[j]     = (bf16_t)((float)s0[j] * (float)d0[j]);
      t[j + 8] = (bf16_t)((float)s1[j] * (float)d1[j]);
    }
    a[ks] = t;
  }
  // ---- e_vec tiles via WMMA
  v8f acc[8];
#pragma unroll
  for (int nt = 0; nt < 8; ++nt) {
    v8f c = {};
#pragma unroll
    for (int ks = 0; ks < 4; ++ks)
      c = WMMA_BF16(a[ks], wmma_b_frag(sW, lane, nt, ks), c);
    acc[nt] = c;
  }
  // ---- epilogue: e = leaky(dot(w_aout, gelu(acc + P[s] + Q[d])) + b)
  int sA[8], dA[8];
#pragma unroll
  for (int r = 0; r < 8; ++r) { sA[r] = sS[wave][r + 8 * h]; dA[r] = sDd[wave][r + 8 * h]; }
  float part[8] = {0, 0, 0, 0, 0, 0, 0, 0};
#pragma unroll
  for (int nt = 0; nt < 8; ++nt) {
    int n = nt * 16 + ln;
    float wa = w_aout[n];
#pragma unroll
    for (int r = 0; r < 8; ++r) {
      float ev = acc[nt][r] + P[(size_t)sA[r] * D + n] + Q[(size_t)dA[r] * D + n];
      part[r] += gelu_f(ev) * wa;
    }
  }
  float ba = b_aout[0];
  float esc[8];
#pragma unroll
  for (int r = 0; r < 8; ++r) {
    float s = part[r];                 // reduce over n across the 16-lane half
    s += __shfl_xor(s, 8);
    s += __shfl_xor(s, 4);
    s += __shfl_xor(s, 2);
    s += __shfl_xor(s, 1);
    float t = s + ba;
    esc[r] = t > 0.0f ? t : 0.2f * t;  // leaky_relu(0.2)
  }
  // ---- scatter: segment_max(e*Hp[src]) and segment_sum(e*Hp2[src]) into dst
#pragma unroll
  for (int r = 0; r < 8; ++r) {
    int e = ebase + r + 8 * h;
    if (e < E) {
      int sI = sA[r], dI = dA[r];
      float es = esc[r];
#pragma unroll
      for (int j = 0; j < 8; ++j) {
        int col = ln + 16 * j;
        size_t si = (size_t)sI * D + col;
        size_t di = (size_t)dI * D + col;
        atomicMaxF(&nmax[di], es * (float)Hp[si]);
        atomicAdd(&nsum[di], es * (float)Hp2[si]);
      }
    }
  }
}

// ---------------------------------------------------------------- combine: rst = feat@Ws^T + neigh@Wn^T + neigh2@Wn2^T + b

__global__ __launch_bounds__(256) void k_combine(
    const float* __restrict__ feat, const float* __restrict__ nmax,
    const float* __restrict__ nsum, const float* __restrict__ deg,
    const bf16_t* __restrict__ W3,
    const float* __restrict__ bs, const float* __restrict__ bn, const float* __restrict__ bn2,
    float* __restrict__ out, int N) {
  __shared__ __align__(32) bf16_t sW[3 * D * D];   // 96 KB
  {
    const uint4* s = (const uint4*)W3;
    uint4* d = (uint4*)sW;
    for (int i = threadIdx.x; i < (3 * D * D * 2) / 16; i += blockDim.x) d[i] = s[i];
  }
  __syncthreads();
  int wave = threadIdx.x >> 5, lane = threadIdx.x & 31;
  int h = (lane >> 4) & 1, ln = lane & 15;
  int row0 = (blockIdx.x * 8 + wave) * 16;
  if (row0 >= N) return;
  int arow = row0 + ln; if (arow > N - 1) arow = N - 1;
  float dg = deg[arow];
  float f1 = dg > 0.0f ? 1.0f : 0.0f;       // mask for segment_max branch
  float f2 = 1.0f / fmaxf(dg, 1.0f);        // mean for segment_sum branch
  const float* srcs[3] = { feat + (size_t)arow * D, nmax + (size_t)arow * D, nsum + (size_t)arow * D };
  float scl[3] = {1.0f, f1, f2};
  v8f zero = {};
  v8f acc[8];
#pragma unroll
  for (int nt = 0; nt < 8; ++nt) acc[nt] = zero;
#pragma unroll
  for (int si = 0; si < 3; ++si) {
#pragma unroll
    for (int ks = 0; ks < 4; ++ks) {
      v16bf a = a_frag_f32(srcs[si], ks * 32 + h * 8, scl[si], 0);
#pragma unroll
      for (int nt = 0; nt < 8; ++nt)
        acc[nt] = WMMA_BF16(a, wmma_b_frag(sW + si * D * D, lane, nt, ks), acc[nt]);
    }
  }
#pragma unroll
  for (int nt = 0; nt < 8; ++nt) {
    int n = nt * 16 + ln;
    float bb = bs[n] + bn[n] + bn2[n];
#pragma unroll
    for (int r = 0; r < 8; ++r) {
      int row = row0 + r + 8 * h;
      if (row < N) out[(size_t)row * D + n] = acc[nt][r] + bb;
    }
  }
}

// ---------------------------------------------------------------- residual MLP layer: rst += gelu(rst)@W^T + b (in place)

__global__ __launch_bounds__(256) void k_mlp(float* __restrict__ rst, const bf16_t* __restrict__ W,
                                             const float* __restrict__ b, int N) {
  __shared__ __align__(32) bf16_t sW[D * D];
  {
    const uint4* s = (const uint4*)W;
    uint4* d = (uint4*)sW;
    for (int i = threadIdx.x; i < (D * D * 2) / 16; i += blockDim.x) d[i] = s[i];
  }
  __syncthreads();
  int wave = threadIdx.x >> 5, lane = threadIdx.x & 31;
  int h = (lane >> 4) & 1, ln = lane & 15;
  int row0 = (blockIdx.x * 8 + wave) * 16;
  if (row0 >= N) return;
  int arow = row0 + ln; if (arow > N - 1) arow = N - 1;
  const float* rowp = rst + (size_t)arow * D;
  v16bf a[4];                          // A = gelu(rst): read fully before stores
#pragma unroll
  for (int ks = 0; ks < 4; ++ks) a[ks] = a_frag_f32(rowp, ks * 32 + h * 8, 1.0f, 1);
#pragma unroll
  for (int nt = 0; nt < 8; ++nt) {
    v8f acc = {};
#pragma unroll
    for (int ks = 0; ks < 4; ++ks)
      acc = WMMA_BF16(a[ks], wmma_b_frag(sW, lane, nt, ks), acc);
    int n = nt * 16 + ln;
    float bv = b[n];
#pragma unroll
    for (int r = 0; r < 8; ++r) {
      int row = row0 + r + 8 * h;
      if (row < N) {
        size_t idx = (size_t)row * D + n;
        rst[idx] = rst[idx] + acc[r] + bv;   // each (row,n) owned by one lane
      }
    }
  }
}

// ---------------------------------------------------------------- host launcher

extern "C" void kernel_launch(void* const* d_in, const int* in_sizes, int n_in,
                              void* d_out, int out_size, void* d_ws, size_t ws_size,
                              hipStream_t stream) {
  (void)n_in; (void)out_size; (void)ws_size;
  const float* feat    = (const float*)d_in[0];
  const int*   src     = (const int*)d_in[1];
  const int*   dst     = (const int*)d_in[2];
  const float* W_asrc  = (const float*)d_in[3];  const float* b_asrc  = (const float*)d_in[4];
  const float* W_adst  = (const float*)d_in[5];  const float* b_adst  = (const float*)d_in[6];
  const float* W_asub  = (const float*)d_in[7];  const float* b_asub  = (const float*)d_in[8];
  const float* W_amul  = (const float*)d_in[9];  const float* b_amul  = (const float*)d_in[10];
  const float* W_aout  = (const float*)d_in[11]; const float* b_aout  = (const float*)d_in[12];
  const float* W_pool  = (const float*)d_in[13]; const float* b_pool  = (const float*)d_in[14];
  const float* W_pool2 = (const float*)d_in[15]; const float* b_pool2 = (const float*)d_in[16];
  const float* W_self  = (const float*)d_in[17]; const float* b_self  = (const float*)d_in[18];
  const float* W_neigh = (const float*)d_in[19]; const float* b_neigh = (const float*)d_in[20];
  const float* W_neigh2= (const float*)d_in[21]; const float* b_neigh2= (const float*)d_in[22];
  const float* W_mlp   = (const float*)d_in[23]; const float* b_mlp   = (const float*)d_in[24];

  const int N = in_sizes[0] / D;
  const int E = in_sizes[1];
  float* out = (float*)d_out;

  // carve workspace (256B-aligned chunks)
  char* wsB = (char*)d_ws;
  size_t off = 0;
  auto carve = [&](size_t bytes) -> char* {
    char* p = wsB + off;
    off += (bytes + 255) & ~(size_t)255;
    return p;
  };
  bf16_t* wts   = (bf16_t*)carve((size_t)10 * D * D * sizeof(bf16_t));
  float*  biasP = (float*) carve(D * sizeof(float));
  bf16_t* featb = (bf16_t*)carve((size_t)N * D * sizeof(bf16_t));
  float*  P     = (float*) carve((size_t)N * D * sizeof(float));
  float*  Q     = (float*) carve((size_t)N * D * sizeof(float));
  bf16_t* Hp    = (bf16_t*)carve((size_t)N * D * sizeof(bf16_t));
  bf16_t* Hp2   = (bf16_t*)carve((size_t)N * D * sizeof(bf16_t));
  float*  nmax  = (float*) carve((size_t)N * D * sizeof(float));
  float*  nsum  = (float*) carve((size_t)N * D * sizeof(float));
  float*  deg   = (float*) carve((size_t)N * sizeof(float));

  const int nbElem = (N * D + 255) / 256;
  const int nbN    = (N + 127) / 128;   // 8 waves x 16 rows per block
  const int nbE    = (E + 127) / 128;   // 8 waves x 16 edges per block

  k_prep<<<(D * D + 255) / 256, 256, 0, stream>>>(
      W_asrc, W_adst, W_asub, W_amul, W_pool, W_pool2, W_self, W_neigh, W_neigh2, W_mlp,
      b_asrc, b_adst, b_asub, b_amul, wts, biasP);
  k_featb<<<nbElem, 256, 0, stream>>>(feat, featb, N * D);
  k_init<<<nbElem, 256, 0, stream>>>(nmax, nsum, deg, N);

  // node precompute GEMMs
  k_gemm_node<<<nbN, 256, 0, stream>>>(feat, wts + 0 * D * D, biasP,   0, 0, P,  nullptr, N);
  k_gemm_node<<<nbN, 256, 0, stream>>>(feat, wts + 1 * D * D, nullptr, 0, 0, Q,  nullptr, N);
  k_gemm_node<<<nbN, 256, 0, stream>>>(feat, wts + 2 * D * D, b_pool,  1, 1, nullptr, Hp,  N);
  k_gemm_node<<<nbN, 256, 0, stream>>>(feat, wts + 3 * D * D, b_pool2, 1, 1, nullptr, Hp2, N);

  // edge attention + scatter
  k_edge<<<nbE, 256, 0, stream>>>(featb, src, dst, P, Q, Hp, Hp2, W_aout, b_aout,
                                  wts + 4 * D * D, nmax, nsum, deg, E);

  // combine (K = 3*128 fused GEMM)
  k_combine<<<nbN, 256, 0, stream>>>(feat, nmax, nsum, deg, wts + 5 * D * D,
                                     b_self, b_neigh, b_neigh2, out, N);

  // two residual MLP layers, in place
  k_mlp<<<nbN, 256, 0, stream>>>(out, wts + 8 * D * D, b_mlp,     N);
  k_mlp<<<nbN, 256, 0, stream>>>(out, wts + 9 * D * D, b_mlp + D, N);
}
